// SAGEGRU_79405355369204
// MI455X (gfx1250) — matrix operand order repaired and checked by reading, not compiled, not run on hardware
//
#include <hip/hip_runtime.h>
#include <hip/hip_bf16.h>

typedef __attribute__((ext_vector_type(16))) _Float16 v16h;
typedef __attribute__((ext_vector_type(8)))  float    v8f;

#define NN 20000      // nodes
#define NE 320000     // edges
#define TT 16         // timesteps
#define RBc 1250      // row blocks of 16 (20000/16)
#define HGc 64
#define HTc 128

#define WIH_HALVES (2 * 24 * 512)   // 24576 halves = 48 KB
#define WHH_HALVES (4 * 24 * 512)   // 49152 halves = 96 KB
#define GRU_LDS_BYTES ((WIH_HALVES + WHH_HALVES) * 2)  // 147456 B

// ---------------------------------------------------------------- utilities

__global__ void k_fill0(float* __restrict__ p, long n) {
  for (long i = (long)blockIdx.x * blockDim.x + threadIdx.x; i < n;
       i += (long)gridDim.x * blockDim.x)
    p[i] = 0.f;
}

__device__ __forceinline__ float red16(float v) {
#pragma unroll
  for (int m = 1; m < 16; m <<= 1) v += __shfl_xor(v, m, 32);
  return v;
}

// fast sigmoid/tanh: v_rcp_f32 instead of correctly-rounded divide
__device__ __forceinline__ float sigm(float x) {
  return __builtin_amdgcn_rcpf(1.f + __expf(-x));
}
__device__ __forceinline__ float tanh_fast(float x) {
  float e = __expf(-2.f * fabsf(x));                    // in (0,1]
  float r = (1.f - e) * __builtin_amdgcn_rcpf(1.f + e); // |tanh|
  return copysignf(r, x);
}

// A-fragment K map for V_WMMA_F32_16X16X32_F16 (16x32 f16 A):
//   lane l (row = l&15), half h in [0,16):  k = 2*(h&8) + 8*(l>=16) + (h&7)
__device__ __forceinline__ int kmapA(int lane, int h) {
  return ((h & 8) << 1) + ((lane >> 4) << 3) + (h & 7);
}

// ---------------------------------------------------------------- graph prep

__global__ void k_deg(const int* __restrict__ ei, float* __restrict__ deg) {
  int e = blockIdx.x * blockDim.x + threadIdx.x;
  if (e < NE) atomicAdd(&deg[ei[NE + e]], 1.f);
}

__global__ void k_scatter0(const int* __restrict__ ei, const float* __restrict__ x,
                           float* __restrict__ agg0) {
  long i = (long)blockIdx.x * blockDim.x + threadIdx.x;
  if (i >= (long)TT * NE) return;
  int t = (int)(i / NE), e = (int)(i % NE);
  atomicAdd(&agg0[t * NN + ei[NE + e]], x[(long)t * NN + ei[e]]);
}

// SAGE layer 0 (IN_FEATS=1 -> rank-1 outer product) + LN + ReLU, per timestep t.
// Output h0 as f16 row-major (N x 64).
__global__ void k_sage0(int t, const float* __restrict__ x, const float* __restrict__ agg0,
                        const float* __restrict__ deg,
                        const float* __restrict__ Wl0, const float* __restrict__ Wr0,
                        const float* __restrict__ b0,
                        const float* __restrict__ g, const float* __restrict__ b,
                        _Float16* __restrict__ h0) {
  int n = blockIdx.x * blockDim.x + threadIdx.x;
  if (n >= NN) return;
  float a = agg0[t * NN + n] / fmaxf(deg[n], 1.f);   // one precise divide / thread
  float xv = x[(long)t * NN + n];
  float v[HGc];
  float mu = 0.f;
#pragma unroll
  for (int f = 0; f < HGc; f++) { v[f] = a * Wl0[f] + xv * Wr0[f] + b0[f]; mu += v[f]; }
  mu *= (1.f / HGc);
  float var = 0.f;
#pragma unroll
  for (int f = 0; f < HGc; f++) { float dd = v[f] - mu; var += dd * dd; }
  float rs = rsqrtf(var * (1.f / HGc) + 1e-5f);
#pragma unroll
  for (int f = 0; f < HGc; f++) {
    float o = (v[f] - mu) * rs * g[f] + b[f];
    h0[(long)n * HGc + f] = (_Float16)fmaxf(o, 0.f);
  }
}

// Edge scatter for layer 1: agg1[dst][0:64] += h0[src][0:64]; 4 threads/edge.
__global__ void k_scatter1(const int* __restrict__ ei, const _Float16* __restrict__ h0,
                           float* __restrict__ agg1) {
  long i = (long)blockIdx.x * blockDim.x + threadIdx.x;
  if (i >= (long)NE * 4) return;
  int e = (int)(i >> 2), p = (int)(i & 3);
  int s = ei[e], d = ei[NE + e];
  const _Float16* hp = h0 + (long)s * HGc + p * 16;
  float* ap = agg1 + (long)d * HGc + p * 16;
#pragma unroll
  for (int f = 0; f < 16; f++) atomicAdd(&ap[f], (float)hp[f]);
}

// ---------------------------------------------------------------- weight repack
// B fragment layout (32x16 f16 B): lane l col n=l&15; k = 2*(half&8) + 8*(l>=16) + (half&7).
// B stored as [kt][jt][lane][half] contiguous (512 halves per fragment).
__global__ void k_repack(const float* __restrict__ W, _Float16* __restrict__ B,
                         int K, int Ncols, int trans) {
  int jtiles = Ncols >> 4;
  int total = (K >> 5) * jtiles * 512;
  int i = blockIdx.x * blockDim.x + threadIdx.x;
  if (i >= total) return;
  int half = i & 15, lane = (i >> 4) & 31, ft = i >> 9;
  int jt = ft % jtiles, kt = ft / jtiles;
  int j = jt * 16 + (lane & 15);
  int k = kt * 32 + ((half & 8) << 1) + ((lane >> 4) << 3) + (half & 7);
  float w = trans ? W[(long)j * K + k] : W[(long)k * Ncols + j];
  B[i] = (_Float16)w;
}

// ---------------------------------------------------------------- SAGE1 (WMMA)
// out = mean_agg @ Wl1 + h0 @ Wr1 + b1 ; LN ; ReLU ; store H[t] in A-frag order.
__global__ void k_sage1(int t, const float* __restrict__ agg1, const float* __restrict__ deg,
                        const _Float16* __restrict__ h0,
                        const _Float16* __restrict__ WlB, const _Float16* __restrict__ WrB,
                        const float* __restrict__ b1,
                        const float* __restrict__ g, const float* __restrict__ bb,
                        _Float16* __restrict__ Hfrag) {
  int wave = threadIdx.x >> 5, lane = threadIdx.x & 31;
  int rb = blockIdx.x * (blockDim.x >> 5) + wave;
  if (rb >= RBc) return;
  int mrow = lane & 15, hi = lane >> 4;
  int row = rb * 16 + mrow;
  float rd = 1.f / fmaxf(deg[row], 1.f);   // one precise divide per lane

  // Build A fragments (2 K-tiles over K=64) for both input matrices.
  v16h aA[2], hA[2];
  const float*    ar = agg1 + (long)row * HGc;
  const _Float16* hr = h0 + (long)row * HGc;
#pragma unroll
  for (int kt = 0; kt < 2; kt++)
#pragma unroll
    for (int h = 0; h < 16; h++) {
      int k = kt * 32 + kmapA(lane, h);
      aA[kt][h] = (_Float16)(ar[k] * rd);
      hA[kt][h] = hr[k];
    }

  const v16h* BL = (const v16h*)WlB;  // [kt*4+jt][lane]
  const v16h* BR = (const v16h*)WrB;
  v8f acc[4];
#pragma unroll
  for (int jt = 0; jt < 4; jt++) {
    v8f c = {};
#pragma unroll
    for (int kt = 0; kt < 2; kt++) {
      c = __builtin_amdgcn_wmma_f32_16x16x32_f16(false, aA[kt], false,
            BL[(kt * 4 + jt) * 32 + lane], (short)0, c, false, false);
      c = __builtin_amdgcn_wmma_f32_16x16x32_f16(false, hA[kt], false,
            BR[(kt * 4 + jt) * 32 + lane], (short)0, c, false, false);
    }
    float bias = b1[jt * 16 + mrow];
#pragma unroll
    for (int e = 0; e < 8; e++) c[e] += bias;
    acc[jt] = c;
  }

  // LayerNorm across the 64 cols of each row (cols live across 16 lanes x 4 tiles).
  float mu[8], rs[8];
#pragma unroll
  for (int e = 0; e < 8; e++) {
    float s = acc[0][e] + acc[1][e] + acc[2][e] + acc[3][e];
    mu[e] = red16(s) * (1.f / HGc);
  }
#pragma unroll
  for (int e = 0; e < 8; e++) {
    float s = 0.f;
#pragma unroll
    for (int jt = 0; jt < 4; jt++) { float dd = acc[jt][e] - mu[e]; s += dd * dd; }
    rs[e] = rsqrtf(red16(s) * (1.f / HGc) + 1e-5f);
  }

  // Scale/shift, ReLU, store as f16 A-fragment layout for the GRU input GEMM.
#pragma unroll
  for (int jt = 0; jt < 4; jt++) {
    int col = jt * 16 + mrow;
    float gg = g[col], b2 = bb[col];
    int kt2 = col >> 5, kp = col & 31;
    int half2 = (kp & 7) + ((kp & 16) ? 8 : 0);
    int laneAdd = (kp & 8) ? 16 : 0;
    _Float16* base = Hfrag + ((size_t)(t * RBc + rb) * 2 + kt2) * 512;
#pragma unroll
    for (int e = 0; e < 8; e++) {
      int m = hi * 8 + e;
      float v = (acc[jt][e] - mu[e]) * rs[e] * gg + b2;
      base[(m + laneAdd) * 16 + half2] = (_Float16)fmaxf(v, 0.f);
    }
  }
}

// ---------------------------------------------------------------- GRU step (WMMA)
// gi = H[t]@Wih^T (K=64), gh = h@Whh^T (K=128); r,z fuse gi+gh in one
// accumulator; n keeps inn / hn separate. 18 WMMAs per 16-col tile, 144/wave.
// Both repacked weight buffers (144 KB) are staged in LDS once per block:
// B fragments then come from ds_load instead of L2 (they exceed the 64 KB WGP$).
__global__ void k_gru(int t, const _Float16* __restrict__ Hfrag,
                      const _Float16* __restrict__ hfP, const float* __restrict__ h32P,
                      const _Float16* __restrict__ WihB, const _Float16* __restrict__ WhhB,
                      const float* __restrict__ bih, const float* __restrict__ bhh,
                      float* __restrict__ h32N, _Float16* __restrict__ hfN) {
  extern __shared__ __align__(32) _Float16 smem[];
  _Float16* sWih = smem;                 // 48 KB
  _Float16* sWhh = smem + WIH_HALVES;    // 96 KB

  // cooperative stage of both weight-fragment buffers into LDS
  {
    const uint4* s0 = (const uint4*)WihB;
    uint4*       d0 = (uint4*)sWih;
    for (int i = threadIdx.x; i < WIH_HALVES / 8; i += blockDim.x) d0[i] = s0[i];
    const uint4* s1 = (const uint4*)WhhB;
    uint4*       d1 = (uint4*)sWhh;
    for (int i = threadIdx.x; i < WHH_HALVES / 8; i += blockDim.x) d1[i] = s1[i];
  }
  __syncthreads();

  int wave = threadIdx.x >> 5, lane = threadIdx.x & 31;
  int rb = blockIdx.x * (blockDim.x >> 5) + wave;
  if (rb >= RBc) return;
  int mrow = lane & 15, hi = lane >> 4;

  // prefetch the f32 h-state rows this wave consumes after its WMMA chain
  __builtin_prefetch(h32P + ((size_t)rb * 16 + mrow) * HTc + hi * 64, 0, 3);

  v16h aH[2], ah[4];
  const v16h* pH = (const v16h*)(Hfrag + (size_t)(t * RBc + rb) * 2 * 512);
  aH[0] = pH[lane]; aH[1] = pH[32 + lane];
  const v16h* ph = (const v16h*)(hfP + (size_t)rb * 4 * 512);
#pragma unroll
  for (int kt = 0; kt < 4; kt++) ah[kt] = ph[kt * 32 + lane];

  const v16h* BI = (const v16h*)sWih;  // [kt*24+jt][lane], 2 K-tiles, 24 j-tiles
  const v16h* BH = (const v16h*)sWhh;  // 4 K-tiles, 24 j-tiles

#pragma unroll 2
  for (int q = 0; q < 8; q++) {        // 16-col tile within each gate
    v8f ar = {}, az = {}, ai = {}, ahn = {};
#pragma unroll
    for (int kt = 0; kt < 2; kt++) {
      ar = __builtin_amdgcn_wmma_f32_16x16x32_f16(false, aH[kt], false,
             BI[(kt * 24 + q) * 32 + lane], (short)0, ar, false, false);
      az = __builtin_amdgcn_wmma_f32_16x16x32_f16(false, aH[kt], false,
             BI[(kt * 24 + 8 + q) * 32 + lane], (short)0, az, false, false);
      ai = __builtin_amdgcn_wmma_f32_16x16x32_f16(false, aH[kt], false,
             BI[(kt * 24 + 16 + q) * 32 + lane], (short)0, ai, false, false);
    }
#pragma unroll
    for (int kt = 0; kt < 4; kt++) {
      ar = __builtin_amdgcn_wmma_f32_16x16x32_f16(false, ah[kt], false,
             BH[(kt * 24 + q) * 32 + lane], (short)0, ar, false, false);
      az = __builtin_amdgcn_wmma_f32_16x16x32_f16(false, ah[kt], false,
             BH[(kt * 24 + 8 + q) * 32 + lane], (short)0, az, false, false);
      ahn = __builtin_amdgcn_wmma_f32_16x16x32_f16(false, ah[kt], false,
             BH[(kt * 24 + 16 + q) * 32 + lane], (short)0, ahn, false, false);
    }
    int jl = q * 16 + mrow;                 // column within HT=128
    float b_ir = bih[jl],           b_hr = bhh[jl];
    float b_iz = bih[HTc + jl],     b_hz = bhh[HTc + jl];
    float b_in = bih[2 * HTc + jl], b_hn = bhh[2 * HTc + jl];
    int kt2 = jl >> 5, kp = jl & 31;
    int half2 = (kp & 7) + ((kp & 16) ? 8 : 0);
    int laneAdd = (kp & 8) ? 16 : 0;
    _Float16* fb = hfN + (size_t)(rb * 4 + kt2) * 512;
#pragma unroll
    for (int e = 0; e < 8; e++) {
      int m = hi * 8 + e;
      long row = (long)rb * 16 + m;
      float hp = h32P[row * HTc + jl];
      float r = sigm(ar[e] + b_ir + b_hr);
      float z = sigm(az[e] + b_iz + b_hz);
      float n = tanh_fast(ai[e] + b_in + r * (ahn[e] + b_hn));
      float hn2 = (1.f - z) * n + z * hp;
      h32N[row * HTc + jl] = hn2;
      fb[(m + laneAdd) * 16 + half2] = (_Float16)hn2;
    }
  }
}

// ---------------------------------------------------------------- head
__global__ void k_head(const float* __restrict__ h32, const float* __restrict__ W,
                       const float* __restrict__ b, float* __restrict__ y) {
  int n = blockIdx.x * blockDim.x + threadIdx.x;
  if (n >= NN) return;
  const float* hr = h32 + (long)n * HTc;
  float s = b[0];
#pragma unroll
  for (int j = 0; j < HTc; j++) s += hr[j] * W[j];
  y[n] = s;
}

// ---------------------------------------------------------------- launch

extern "C" void kernel_launch(void* const* d_in, const int* in_sizes, int n_in,
                              void* d_out, int out_size, void* d_ws, size_t ws_size,
                              hipStream_t stream) {
  const float* x    = (const float*)d_in[0];
  const int*   ei   = (const int*)d_in[1];
  const float* Wl0  = (const float*)d_in[2];
  const float* Wr0  = (const float*)d_in[3];
  const float* b0   = (const float*)d_in[4];
  const float* ln0g = (const float*)d_in[5];
  const float* ln0b = (const float*)d_in[6];
  const float* Wl1  = (const float*)d_in[7];
  const float* Wr1  = (const float*)d_in[8];
  const float* b1   = (const float*)d_in[9];
  const float* ln1g = (const float*)d_in[10];
  const float* ln1b = (const float*)d_in[11];
  const float* Wih  = (const float*)d_in[12];
  const float* Whh  = (const float*)d_in[13];
  const float* bih  = (const float*)d_in[14];
  const float* bhh  = (const float*)d_in[15];
  const float* hW   = (const float*)d_in[16];
  const float* hb   = (const float*)d_in[17];
  float* out = (float*)d_out;
  (void)in_sizes; (void)n_in; (void)out_size; (void)ws_size;

  char* w = (char*)d_ws; size_t off = 0;
  auto alloc = [&](size_t bytes) -> char* {
    char* p = w + off; off = (off + bytes + 511) & ~(size_t)511; return p;
  };
  float*    deg   = (float*)alloc((size_t)NN * 4);
  float*    agg0  = (float*)alloc((size_t)TT * NN * 4);
  _Float16* h0    = (_Float16*)alloc((size_t)NN * HGc * 2);
  float*    agg1  = (float*)alloc((size_t)NN * HGc * 4);
  _Float16* Hfrag = (_Float16*)alloc((size_t)TT * RBc * 2 * 512 * 2);
  float*    h32a  = (float*)alloc((size_t)NN * HTc * 4);
  float*    h32b  = (float*)alloc((size_t)NN * HTc * 4);
  _Float16* hfa   = (_Float16*)alloc((size_t)RBc * 4 * 512 * 2);
  _Float16* hfb   = (_Float16*)alloc((size_t)RBc * 4 * 512 * 2);
  _Float16* WihB  = (_Float16*)alloc((size_t)WIH_HALVES * 2);
  _Float16* WhhB  = (_Float16*)alloc((size_t)WHH_HALVES * 2);
  _Float16* WlB   = (_Float16*)alloc((size_t)2 * 4 * 512 * 2);
  _Float16* WrB   = (_Float16*)alloc((size_t)2 * 4 * 512 * 2);

  // zero accumulators / initial GRU state (every call; graph-replay safe)
  k_fill0<<<512, 256, 0, stream>>>(deg, NN);
  k_fill0<<<512, 256, 0, stream>>>(agg0, (long)TT * NN);
  k_fill0<<<512, 256, 0, stream>>>(h32a, (long)NN * HTc);
  k_fill0<<<512, 256, 0, stream>>>((float*)hfa, (long)RBc * 4 * 512 / 2);

  k_deg<<<(NE + 255) / 256, 256, 0, stream>>>(ei, deg);
  k_scatter0<<<(int)(((long)TT * NE + 255) / 256), 256, 0, stream>>>(ei, x, agg0);

  k_repack<<<(WIH_HALVES + 255) / 256, 256, 0, stream>>>(Wih, WihB, 64, 384, 1);
  k_repack<<<(WHH_HALVES + 255) / 256, 256, 0, stream>>>(Whh, WhhB, 128, 384, 1);
  k_repack<<<(2 * 4 * 512 + 255) / 256, 256, 0, stream>>>(Wl1, WlB, 64, 64, 0);
  k_repack<<<(2 * 4 * 512 + 255) / 256, 256, 0, stream>>>(Wr1, WrB, 64, 64, 0);

  for (int t = 0; t < TT; t++) {
    k_sage0<<<(NN + 255) / 256, 256, 0, stream>>>(t, x, agg0, deg, Wl0, Wr0, b0,
                                                  ln0g, ln0b, h0);
    k_fill0<<<512, 256, 0, stream>>>(agg1, (long)NN * HGc);
    k_scatter1<<<(NE * 4 + 255) / 256, 256, 0, stream>>>(ei, h0, agg1);
    k_sage1<<<(RBc + 7) / 8, 256, 0, stream>>>(t, agg1, deg, h0, WlB, WrB, b1,
                                               ln1g, ln1b, Hfrag);
  }

  for (int t = 0; t < TT; t++) {
    const float*    hp = (t & 1) ? h32b : h32a;
    float*          hn = (t & 1) ? h32a : h32b;
    const _Float16* fp = (t & 1) ? hfb : hfa;
    _Float16*       fn = (t & 1) ? hfa : hfb;
    k_gru<<<(RBc + 7) / 8, 256, GRU_LDS_BYTES, stream>>>(t, Hfrag, fp, hp, WihB, WhhB,
                                                         bih, bhh, hn, fn);
  }
  // t=15 (odd) wrote h32a
  k_head<<<(NN + 255) / 256, 256, 0, stream>>>(h32a, hW, hb, out);
}
